// CausalSelfAttention_52063593562516
// MI455X (gfx1250) — matrix operand order repaired and checked by reading, not compiled, not run on hardware
//
#include <hip/hip_runtime.h>
#include <hip/hip_bf16.h>

typedef __attribute__((ext_vector_type(16))) _Float16 v16h;
typedef __attribute__((ext_vector_type(8)))  _Float16 v8h;
typedef __attribute__((ext_vector_type(4)))  _Float16 v4h;
typedef __attribute__((ext_vector_type(8)))  float    v8f;
typedef __attribute__((ext_vector_type(4)))  float    v4f;
typedef __attribute__((ext_vector_type(4)))  int      v4i;

#define Bsz   2
#define Tseq  2048
#define Cdim  1024
#define Hh    16
#define HD    64
#define Mrows (Bsz * Tseq)        // 4096

#define BSTRIDE 40   // LDS row stride in halves: 80B rows -> 16B aligned, conflict-free b128

// ---- CDNA5 async global->LDS copy (ASYNCcnt path) ----
typedef __attribute__((address_space(3))) v4i* lds_v4i_ptr;

#if __has_builtin(__builtin_amdgcn_global_load_async_to_lds_b128)
#define ASYNC_LDS_B128(gp, lp)                                                  \
  __builtin_amdgcn_global_load_async_to_lds_b128(                               \
      (v4i*)(gp),                                                               \
      (lds_v4i_ptr)(unsigned int)(uintptr_t)(lp),                               \
      0, 0)
#else
#define ASYNC_LDS_B128(gp, lp)                                                  \
  asm volatile("global_load_async_to_lds_b128 %0, %1, off"                      \
               :: "v"((unsigned int)(uintptr_t)(lp)), "v"(gp) : "memory")
#endif

#if __has_builtin(__builtin_amdgcn_s_wait_asynccnt)
#define WAIT_ASYNC0() __builtin_amdgcn_s_wait_asynccnt(0)
#else
#define WAIT_ASYNC0() asm volatile("s_wait_asynccnt 0" ::: "memory")
#endif

__device__ __forceinline__ v8f wmma_f16(v16h a, v16h b, v8f c) {
  return __builtin_amdgcn_wmma_f32_16x16x32_f16(
      /*neg_a=*/false, a, /*neg_b=*/false, b,
      /*c_mod=*/(short)0, c, /*reuse_a=*/false, /*reuse_b=*/false);
}

// A-fragment (16x32 f16, MxK), A row-major at `base` (tile origin), leading dim ld.
// Lane l: row M=l&15; elems 0-7 = K g*8+0..7, elems 8-15 = K 16+g*8+0..7 (g=l>>4).
__device__ __forceinline__ v16h load_a16(const _Float16* base, int ld) {
  const int l = threadIdx.x & 31;
  const int r = l & 15, g = l >> 4;
  const _Float16* p = base + r * ld + g * 8;
  v8h lo = *(const v8h*)p;
  v8h hi = *(const v8h*)(p + 16);
  return __builtin_shufflevector(lo, hi, 0,1,2,3,4,5,6,7,8,9,10,11,12,13,14,15);
}

// B-fragment (32x16 f16, KxN) given B^T row-major: baseT at (n0,k0), row n contiguous in k.
// Lane l: col N=l&15; elems 0-7 = K g*8+0..7, elems 8-15 = K 16+g*8+0..7 (g=l>>4).
__device__ __forceinline__ v16h load_b16(const _Float16* baseT, int ld) {
  const int l = threadIdx.x & 31;
  const int r = l & 15, g = l >> 4;
  const _Float16* p = baseT + r * ld + g * 8;
  v8h lo = *(const v8h*)p;
  v8h hi = *(const v8h*)(p + 16);
  return __builtin_shufflevector(lo, hi, 0,1,2,3,4,5,6,7,8,9,10,11,12,13,14,15);
}

// ---------------------------------------------------------------- f32 -> f16
__global__ void __launch_bounds__(256)
cvt_f32_f16_kernel(const float* __restrict__ src, _Float16* __restrict__ dst, int n4) {
  int i = blockIdx.x * blockDim.x + threadIdx.x;
  if (i < n4) {
    v4f v = ((const v4f*)src)[i];
    v4h h;
    h[0] = (_Float16)v[0]; h[1] = (_Float16)v[1];
    h[2] = (_Float16)v[2]; h[3] = (_Float16)v[3];
    ((v4h*)dst)[i] = h;
  }
}

// ------------------------------------------------------ block-cooperative GEMM core
// Block = 8 waves; each wave owns a 32-row m-tile (2 subtiles); all share a 64-col n-tile.
// B tile (64 x 32 halves) staged into double-buffered LDS via async copies.
// Per wave: acc[8] = 32x64 f32 output tile (acc[0..3]: rows 0-15, acc[4..7]: rows 16-31).
__device__ __forceinline__ void gemm_tile_32x64(
    const _Float16* __restrict__ arow,     // A + m0*Cdim (this wave's 32 rows)
    const _Float16* __restrict__ wtile,    // W + n0*Cdim
    _Float16 (*bbuf)[64 * BSTRIDE],        // [2][64*BSTRIDE] shared
    v8f acc[8]) {
  const int tid = threadIdx.x;             // 256 threads
  const int bn  = tid >> 2;                // 0..63 : B row (n)
  const int bk  = (tid & 3) * 8;           // 0,8,16,24 : k-chunk (halves)

  // prefetch K-step 0
  ASYNC_LDS_B128(wtile + (size_t)bn * Cdim + bk, &bbuf[0][bn * BSTRIDE + bk]);
  v16h a0 = load_a16(arow, Cdim);
  v16h a1 = load_a16(arow + (size_t)16 * Cdim, Cdim);

  int cur = 0;
  for (int k0 = 0; k0 < Cdim; k0 += 32) {
    WAIT_ASYNC0();          // my async copy into bbuf[cur] done
    __syncthreads();        // everyone's copy done; prior reads of bbuf[cur^1] done
    const bool more = (k0 + 32 < Cdim);
    if (more)
      ASYNC_LDS_B128(wtile + (size_t)bn * Cdim + (k0 + 32) + bk,
                     &bbuf[cur ^ 1][bn * BSTRIDE + bk]);
    v16h a0n = a0, a1n = a1;
    if (more) {
      a0n = load_a16(arow + k0 + 32, Cdim);
      a1n = load_a16(arow + (size_t)16 * Cdim + k0 + 32, Cdim);
    }

    const _Float16* bb = &bbuf[cur][0];
    v16h b0 = load_b16(bb +  0 * BSTRIDE, BSTRIDE);
    v16h b1 = load_b16(bb + 16 * BSTRIDE, BSTRIDE);
    v16h b2 = load_b16(bb + 32 * BSTRIDE, BSTRIDE);
    v16h b3 = load_b16(bb + 48 * BSTRIDE, BSTRIDE);
    acc[0] = wmma_f16(a0, b0, acc[0]);
    acc[4] = wmma_f16(a1, b0, acc[4]);
    acc[1] = wmma_f16(a0, b1, acc[1]);
    acc[5] = wmma_f16(a1, b1, acc[5]);
    acc[2] = wmma_f16(a0, b2, acc[2]);
    acc[6] = wmma_f16(a1, b2, acc[6]);
    acc[3] = wmma_f16(a0, b3, acc[3]);
    acc[7] = wmma_f16(a1, b3, acc[7]);

    a0 = a0n;
    a1 = a1n;
    cur ^= 1;
  }
}

// ---------------------------------------------------------------- QKV GEMM
// qkv[m,n] = sum_k xh[m,k]*wqkv[n,k]; M=4096,N=3072,K=1024.
// Grid: (M/256)*(N/64) = 16*48 = 768 blocks. Scatter q/k -> [b,h,t,d], v -> Vt[b,h,d,t].
__global__ void __launch_bounds__(256)
qkv_gemm_kernel(const _Float16* __restrict__ xh, const _Float16* __restrict__ wh,
                _Float16* __restrict__ Qb, _Float16* __restrict__ Kb,
                _Float16* __restrict__ Vt) {
  __shared__ __align__(16) _Float16 bbuf[2][64 * BSTRIDE];
  const int NT64 = 3072 / 64;
  const int w  = threadIdx.x >> 5;
  const int l  = threadIdx.x & 31;
  const int g  = l >> 4;
  const int m0 = (blockIdx.x / NT64) * 256 + w * 32;
  const int n0 = (blockIdx.x % NT64) * 64;

  v8f acc[8] = {v8f{}, v8f{}, v8f{}, v8f{}, v8f{}, v8f{}, v8f{}, v8f{}};
  gemm_tile_32x64(xh + (size_t)m0 * Cdim, wh + (size_t)n0 * Cdim, bbuf, acc);

#pragma unroll
  for (int mt = 0; mt < 2; ++mt) {
#pragma unroll
    for (int j = 0; j < 4; ++j) {
      const int col = n0 + 16 * j + (l & 15);
      const int sec = col >> 10;          // 0=q, 1=k, 2=v
      const int cc  = col & 1023;
      const int h   = cc >> 6;
      const int d   = cc & 63;
#pragma unroll
      for (int i = 0; i < 8; ++i) {
        const int m = m0 + mt * 16 + i + 8 * g;
        const int b = m >> 11;
        const int t = m & 2047;
        _Float16 val = (_Float16)acc[mt * 4 + j][i];
        if (sec == 0)      Qb[(((size_t)(b * Hh + h) * Tseq + t) << 6) + d] = val;
        else if (sec == 1) Kb[(((size_t)(b * Hh + h) * Tseq + t) << 6) + d] = val;
        else               Vt[(((size_t)(b * Hh + h) * HD + d) << 11) + t] = val;
      }
    }
  }
}

// ---------------------------------------------------------------- Flash attention
// One wave per 16-query tile of one (b,h); online softmax over 32-key tiles.
#define PSTRIDE 40
__global__ void __launch_bounds__(256)
attn_kernel(const _Float16* __restrict__ Qb, const _Float16* __restrict__ Kb,
            const _Float16* __restrict__ Vt, _Float16* __restrict__ Y) {
  __shared__ __align__(16) _Float16 pbuf[8][16 * PSTRIDE];

  const int w    = threadIdx.x >> 5;
  const int l    = threadIdx.x & 31;
  const int g    = l >> 4;
  const int tile = blockIdx.x * 8 + w;
  const int bh   = tile >> 7;           // (b*16 + h)
  const int q0   = (tile & 127) * 16;

  const _Float16* Qp = Qb + ((size_t)bh << 17);
  const _Float16* Kp = Kb + ((size_t)bh << 17);
  const _Float16* Vp = Vt + ((size_t)bh << 17);       // [64][2048]
  _Float16* pb = &pbuf[w][0];

  v16h qf0 = load_a16(Qp + (size_t)q0 * HD +  0, HD);
  v16h qf1 = load_a16(Qp + (size_t)q0 * HD + 32, HD);

  v8f o0 = {}, o1 = {}, o2 = {}, o3 = {};
  float mrow[8], lrow[8];
#pragma unroll
  for (int i = 0; i < 8; ++i) { mrow[i] = -1e30f; lrow[i] = 0.f; }

  const int nkt = (q0 + 47) >> 5;
  for (int kt = 0; kt < nkt; ++kt) {
    const int kv0 = kt * 32;

    v8f s0 = {}, s1 = {};
    {
      v16h ka0 = load_b16(Kp + (size_t)(kv0 +  0) * HD +  0, HD);
      v16h ka1 = load_b16(Kp + (size_t)(kv0 +  0) * HD + 32, HD);
      v16h kb0 = load_b16(Kp + (size_t)(kv0 + 16) * HD +  0, HD);
      v16h kb1 = load_b16(Kp + (size_t)(kv0 + 16) * HD + 32, HD);
      s0 = wmma_f16(qf0, ka0, s0);
      s0 = wmma_f16(qf1, ka1, s0);
      s1 = wmma_f16(qf0, kb0, s1);
      s1 = wmma_f16(qf1, kb1, s1);
    }

    const bool need_mask = (kt == nkt - 1);
    float p0a[8], p1a[8], scl[8];
#pragma unroll
    for (int i = 0; i < 8; ++i) {
      float a = s0[i] * 0.125f;
      float b = s1[i] * 0.125f;
      if (need_mask) {
        const int query = q0 + i + 8 * g;
        const int key0  = kv0 + (l & 15);
        a = (key0      > query) ? -1e30f : a;
        b = (key0 + 16 > query) ? -1e30f : b;
      }
      float mx = fmaxf(a, b);
      mx = fmaxf(mx, __shfl_xor(mx, 1, 32));
      mx = fmaxf(mx, __shfl_xor(mx, 2, 32));
      mx = fmaxf(mx, __shfl_xor(mx, 4, 32));
      mx = fmaxf(mx, __shfl_xor(mx, 8, 32));
      const float mnew = fmaxf(mrow[i], mx);
      const float sc   = __expf(mrow[i] - mnew);
      mrow[i] = mnew;
      const float p0 = __expf(a - mnew);
      const float p1 = __expf(b - mnew);
      float rs = p0 + p1;
      rs += __shfl_xor(rs, 1, 32);
      rs += __shfl_xor(rs, 2, 32);
      rs += __shfl_xor(rs, 4, 32);
      rs += __shfl_xor(rs, 8, 32);
      lrow[i] = lrow[i] * sc + rs;
      scl[i] = sc;
      p0a[i] = p0;
      p1a[i] = p1;
    }
#pragma unroll
    for (int i = 0; i < 8; ++i) {
      o0[i] *= scl[i]; o1[i] *= scl[i]; o2[i] *= scl[i]; o3[i] *= scl[i];
    }

    // P (C-layout) -> LDS (row-major 16x32) -> A-fragment
#pragma unroll
    for (int i = 0; i < 8; ++i) {
      const int row = i + 8 * g;
      pb[row * PSTRIDE + (l & 15)]      = (_Float16)p0a[i];
      pb[row * PSTRIDE + (l & 15) + 16] = (_Float16)p1a[i];
    }
    asm volatile("s_wait_dscnt 0" ::: "memory");
    v16h pf = load_a16(pb, PSTRIDE);

    v16h vf0 = load_b16(Vp + (size_t)( 0) * Tseq + kv0, Tseq);
    v16h vf1 = load_b16(Vp + (size_t)(16) * Tseq + kv0, Tseq);
    v16h vf2 = load_b16(Vp + (size_t)(32) * Tseq + kv0, Tseq);
    v16h vf3 = load_b16(Vp + (size_t)(48) * Tseq + kv0, Tseq);
    o0 = wmma_f16(pf, vf0, o0);
    o1 = wmma_f16(pf, vf1, o1);
    o2 = wmma_f16(pf, vf2, o2);
    o3 = wmma_f16(pf, vf3, o3);
  }

  const int b = bh >> 4;
  const int h = bh & 15;
  v8f oacc[4] = {o0, o1, o2, o3};
#pragma unroll
  for (int i = 0; i < 8; ++i) {
    const float inv = 1.0f / lrow[i];
    const int t = q0 + i + 8 * g;
    _Float16* yp = Y + ((size_t)(b * Tseq + t) * Cdim) + h * HD;
#pragma unroll
    for (int d = 0; d < 4; ++d)
      yp[d * 16 + (l & 15)] = (_Float16)(oacc[d][i] * inv);
  }
}

// ---------------------------------------------------------------- Projection GEMM
// out[m,n] = sum_k Y[m,k]*wproj[n,k]; M=4096,N=1024,K=1024; f32 out.
// Grid: 16*16 = 256 blocks.
__global__ void __launch_bounds__(256)
proj_gemm_kernel(const _Float16* __restrict__ Yh, const _Float16* __restrict__ wh,
                 float* __restrict__ out) {
  __shared__ __align__(16) _Float16 bbuf[2][64 * BSTRIDE];
  const int NT64 = Cdim / 64;
  const int w  = threadIdx.x >> 5;
  const int l  = threadIdx.x & 31;
  const int g  = l >> 4;
  const int m0 = (blockIdx.x / NT64) * 256 + w * 32;
  const int n0 = (blockIdx.x % NT64) * 64;

  v8f acc[8] = {v8f{}, v8f{}, v8f{}, v8f{}, v8f{}, v8f{}, v8f{}, v8f{}};
  gemm_tile_32x64(Yh + (size_t)m0 * Cdim, wh + (size_t)n0 * Cdim, bbuf, acc);

#pragma unroll
  for (int mt = 0; mt < 2; ++mt) {
#pragma unroll
    for (int j = 0; j < 4; ++j) {
      const int col = n0 + 16 * j + (l & 15);
#pragma unroll
      for (int i = 0; i < 8; ++i) {
        const int m = m0 + mt * 16 + i + 8 * g;
        out[(size_t)m * Cdim + col] = acc[mt * 4 + j][i];
      }
    }
  }
}

// ---------------------------------------------------------------- launch
extern "C" void kernel_launch(void* const* d_in, const int* in_sizes, int n_in,
                              void* d_out, int out_size, void* d_ws, size_t ws_size,
                              hipStream_t stream) {
  const float* x     = (const float*)d_in[0];   // [2,2048,1024]
  const float* wqkv  = (const float*)d_in[1];   // [3072,1024]
  const float* wproj = (const float*)d_in[2];   // [1024,1024]
  float* out = (float*)d_out;

  char* ws = (char*)d_ws;
  _Float16* xh    = (_Float16*)(ws);                         // 8 MB
  _Float16* wqkvh = (_Float16*)(ws + (8u  << 20));           // 6 MB
  _Float16* wprjh = (_Float16*)(ws + (14u << 20));           // 2 MB
  _Float16* Qb    = (_Float16*)(ws + (16u << 20));           // 8 MB
  _Float16* Kb    = (_Float16*)(ws + (24u << 20));           // 8 MB
  _Float16* Vt    = (_Float16*)(ws + (32u << 20));           // 8 MB
  _Float16* Yh    = (_Float16*)(ws + (40u << 20));           // 8 MB

  cvt_f32_f16_kernel<<<4096, 256, 0, stream>>>(x,     xh,    (Bsz*Tseq*Cdim) / 4);
  cvt_f32_f16_kernel<<<3072, 256, 0, stream>>>(wqkv,  wqkvh, (3*Cdim*Cdim)   / 4);
  cvt_f32_f16_kernel<<<1024, 256, 0, stream>>>(wproj, wprjh, (Cdim*Cdim)     / 4);

  qkv_gemm_kernel<<<768, 256, 0, stream>>>(xh, wqkvh, Qb, Kb, Vt);
  attn_kernel<<<512, 256, 0, stream>>>(Qb, Kb, Vt, Yh);
  proj_gemm_kernel<<<256, 256, 0, stream>>>(Yh, wprjh, out);
}
